// DCMHAttention_12756052869652
// MI455X (gfx1250) — matrix operand order repaired and checked by reading, not compiled, not run on hardware
//
#include <hip/hip_runtime.h>
#include <hip/hip_bf16.h>
#include <stdint.h>

// Problem constants (from reference)
#define B_   2
#define T_   1024
#define D_   2048
#define H_   16
#define HD_  128
#define NEGV (-65504.0f)
#define EPS_ (1e-6f)

typedef unsigned int   u32;
typedef unsigned short u16;
typedef __attribute__((ext_vector_type(16))) __bf16 v16bf;
typedef __attribute__((ext_vector_type(8)))  float  v8f;
typedef __attribute__((ext_vector_type(4)))  u32    u32x4;

// ---------------- bf16 conversion (RNE) ----------------
__device__ __forceinline__ u16 f2bf(float f) {
  union { float f; u32 u; } c; c.f = f;
  u32 u = c.u;
  u += 0x7FFFu + ((u >> 16) & 1u);
  return (u16)(u >> 16);
}

// ---------------- WMMA fragment loaders ----------------
// A: 16x32 bf16 tile from row-major A[M,K].
// Lane L<16 : row m0+L, K = {k0..k0+15}
// Lane L>=16: row m0+L-16, K = {k0+16..k0+31}
__device__ __forceinline__ v16bf load_a_frag(const u16* __restrict__ p /* = &A[(m0+lm)*lda + k0 + hi*16] */) {
  union { v16bf v; u32x4 q[2]; } u;
  u.q[0] = *(const u32x4*)(p);
  u.q[1] = *(const u32x4*)(p + 8);
  return u.v;
}
// B: 32x16 bf16 tile from Bt[N,K] row-major (i.e. B column-major).
// Lane L<16 : col n0+L, K = {k0..k0+15}; lane L>=16: col n0+L-16, K = {k0+16..k0+31}
__device__ __forceinline__ v16bf load_b_frag(const u16* __restrict__ p /* = &Bt[(n0+ln)*ldb + k0 + hi*16] */) {
  union { v16bf v; u32x4 q[2]; } u;
  u.q[0] = *(const u32x4*)(p);
  u.q[1] = *(const u32x4*)(p + 8);
  return u.v;
}

#define WMMA_BF16(ACC, AF, BF) \
  ACC = __builtin_amdgcn_wmma_f32_16x16x32_bf16(false, AF, false, BF, (short)0, ACC, false, false)

// ---------------- Batched GEMM: C[M,N] (f32) = A[M,K](bf16) * Bt[N,K](bf16)^T ----------------
// block = 128 threads (4 waves). Each wave owns a 32x64 output tile:
// 2 A-fragments x 4 B-fragments -> 8 WMMA per k-step, B-fragment reused across both A rows.
__global__ __launch_bounds__(128) void gemm_bt_kernel(
    const u16* __restrict__ A, const u16* __restrict__ Bt, float* __restrict__ C,
    int M, int N, int K, long long sA, long long sB, long long sC)
{
  const int bz = blockIdx.z;
  A  += (long long)bz * sA;
  Bt += (long long)bz * sB;
  C  += (long long)bz * sC;

  const int lane = threadIdx.x & 31;
  const int wave = threadIdx.x >> 5;
  const int m0 = blockIdx.y * 32;
  const int n0 = (blockIdx.x * 4 + wave) * 64;
  if (m0 >= M || n0 >= N) return;

  const int lm = lane & 15;
  const int hi = lane >> 4;

  const u16* aRow0 = A + (long long)(m0 + lm) * K + hi * 16;
  const u16* aRow1 = aRow0 + (long long)16 * K;
  const u16* b0 = Bt + (long long)(n0 +  0 + lm) * K + hi * 16;
  const u16* b1 = Bt + (long long)(n0 + 16 + lm) * K + hi * 16;
  const u16* b2 = Bt + (long long)(n0 + 32 + lm) * K + hi * 16;
  const u16* b3 = Bt + (long long)(n0 + 48 + lm) * K + hi * 16;

  v8f z = {0.f,0.f,0.f,0.f,0.f,0.f,0.f,0.f};
  v8f acc00 = z, acc01 = z, acc02 = z, acc03 = z;
  v8f acc10 = z, acc11 = z, acc12 = z, acc13 = z;

  for (int k0 = 0; k0 < K; k0 += 32) {
    // speculative prefetch of the k-tile two steps ahead (global_prefetch_b8)
    __builtin_prefetch(aRow0 + k0 + 64, 0, 1);
    __builtin_prefetch(aRow1 + k0 + 64, 0, 1);
    __builtin_prefetch(b0    + k0 + 64, 0, 1);

    v16bf a0 = load_a_frag(aRow0 + k0);
    v16bf a1 = load_a_frag(aRow1 + k0);

    v16bf bb = load_b_frag(b0 + k0);
    WMMA_BF16(acc00, a0, bb);
    WMMA_BF16(acc10, a1, bb);
    bb = load_b_frag(b1 + k0);
    WMMA_BF16(acc01, a0, bb);
    WMMA_BF16(acc11, a1, bb);
    bb = load_b_frag(b2 + k0);
    WMMA_BF16(acc02, a0, bb);
    WMMA_BF16(acc12, a1, bb);
    bb = load_b_frag(b3 + k0);
    WMMA_BF16(acc03, a0, bb);
    WMMA_BF16(acc13, a1, bb);
  }

  // C/D layout: VGPR r -> row (tile_m + r + hi*8), col n0 + lm (+ n-tile offset)
  float* crow0 = C + (long long)(m0 + hi * 8) * N + n0 + lm;
  float* crow1 = crow0 + (long long)16 * N;
#pragma unroll
  for (int r = 0; r < 8; ++r) {
    crow0[(long long)r * N +  0] = acc00[r];
    crow0[(long long)r * N + 16] = acc01[r];
    crow0[(long long)r * N + 32] = acc02[r];
    crow0[(long long)r * N + 48] = acc03[r];
    crow1[(long long)r * N +  0] = acc10[r];
    crow1[(long long)r * N + 16] = acc11[r];
    crow1[(long long)r * N + 32] = acc12[r];
    crow1[(long long)r * N + 48] = acc13[r];
  }
}

// ---------------- f32 -> bf16 elementwise ----------------
__global__ void f32_to_bf16_kernel(const float* __restrict__ in, u16* __restrict__ out, int n) {
  int i = blockIdx.x * 256 + threadIdx.x;
  if (i < n) out[i] = f2bf(in[i]);
}

// ---------------- transpose + convert: out[c, r] = bf16(in[r, c]) ----------------
__global__ __launch_bounds__(256) void transpose_bf16_kernel(
    const float* __restrict__ in, u16* __restrict__ out, int R, int Ccol)
{
  __shared__ float tile[32][33];
  int c0 = blockIdx.x * 32, r0 = blockIdx.y * 32;
  int tx = threadIdx.x & 31, ty = threadIdx.x >> 5;
#pragma unroll
  for (int j = 0; j < 32; j += 8)
    tile[ty + j][tx] = in[(long long)(r0 + ty + j) * Ccol + c0 + tx];
  __syncthreads();
#pragma unroll
  for (int j = 0; j < 32; j += 8)
    out[(long long)(c0 + ty + j) * R + r0 + tx] = f2bf(tile[tx][ty + j]);
}

// ---------------- RoPE + scale + pack q/k [b,h,t,hd], v^T [b,h,hd,t] ----------------
__global__ void rope_pack_kernel(const float* __restrict__ qkv, const float* __restrict__ freqs,
                                 u16* __restrict__ qb, u16* __restrict__ kb, u16* __restrict__ vbT)
{
  int idx = blockIdx.x * 256 + threadIdx.x;        // (b,t,h,i) with i < 64
  if (idx >= B_ * T_ * H_ * 64) return;
  int i = idx & 63;
  int h = (idx >> 6) & (H_ - 1);
  int t = (idx >> 10) & (T_ - 1);
  int b = idx >> 20;

  long long base = (long long)(b * T_ + t) * (3 * H_ * HD_);
  float qa = qkv[base + h * HD_ + i];
  float qh = qkv[base + h * HD_ + 64 + i];
  float ka = qkv[base + H_ * HD_ + h * HD_ + i];
  float kh = qkv[base + H_ * HD_ + h * HD_ + 64 + i];
  float cs = freqs[(t * 64 + i) * 2 + 0];
  float sn = freqs[(t * 64 + i) * 2 + 1];
  const float scale = 0.08838834764831845f;        // HD^-0.5

  long long ob = (long long)((b * H_ + h) * T_ + t) * HD_;
  qb[ob + 2 * i    ] = f2bf((qa * cs - qh * sn) * scale);
  qb[ob + 2 * i + 1] = f2bf((qh * cs + qa * sn) * scale);
  kb[ob + 2 * i    ] = f2bf(ka * cs - kh * sn);
  kb[ob + 2 * i + 1] = f2bf(kh * cs + ka * sn);

  float v0 = qkv[base + 2 * H_ * HD_ + h * HD_ + 2 * i];
  float v1 = qkv[base + 2 * H_ * HD_ + h * HD_ + 2 * i + 1];
  long long vb = (long long)(b * H_ + h) * HD_ * T_;
  vbT[vb + (long long)(2 * i    ) * T_ + t] = f2bf(v0);
  vbT[vb + (long long)(2 * i + 1) * T_ + t] = f2bf(v1);
}

// ---------------- dynamic weights: GELU -> qkw projection -> RMS-norm w1, tanh dd ----------------
__global__ __launch_bounds__(128) void dynw_kernel(
    const float* __restrict__ hid, const float* __restrict__ ddpre, const float* __restrict__ qkw,
    float* __restrict__ pw1, float* __restrict__ pw2, float* __restrict__ pdd)
{
  int bt = blockIdx.x;
  int tid = threadIdx.x;
  __shared__ float h[512];
  __shared__ float wbuf[4][2][16];

  for (int j = tid; j < 512; j += 128) {
    float x = hid[(long long)bt * 512 + j];
    h[j] = 0.5f * x * (1.0f + erff(x * 0.70710678118654752f));   // exact GELU
  }
  __syncthreads();

  int c = tid >> 5, iw = (tid >> 4) & 1, m = tid & 15;
  float s = 0.f;
  for (int k = 0; k < 128; ++k)
    s += h[c * 128 + k] * qkw[((c * 128 + k) * 2 + iw) * 16 + m];
  wbuf[c][iw][m] = s;
  __syncthreads();

  if (iw == 0) {
    float ss = 0.f;
#pragma unroll
    for (int j = 0; j < 16; ++j) { float w = wbuf[c][0][j]; ss += w * w; }
    float sc = rsqrtf(ss * (1.0f / 16.0f) + EPS_);
    pw1[((long long)bt * 4 + c) * 16 + m] = wbuf[c][0][m] * sc;
  } else {
    pw2[((long long)bt * 4 + c) * 16 + m] = wbuf[c][1][m];
  }
  if (tid < 64) {
    int cc = tid >> 4, mm = tid & 15;
    pdd[((long long)bt * 4 + cc) * 16 + mm] = tanhf(ddpre[(long long)bt * 64 + cc * 16 + mm]);
  }
}

// ---------------- fused pre-mix + mask + softmax + post-mix (one (b,t) row per block) ----------------
// Param layout per (b,t): c = 0:pre_q, 1:pre_k, 2:post_q, 3:post_k ; 16 floats each.
__global__ __launch_bounds__(256) void mixer_kernel(
    const float* __restrict__ logits,
    const float* __restrict__ pw1, const float* __restrict__ pw2, const float* __restrict__ pdd,
    u16* __restrict__ probsB)
{
  __shared__ float row[H_ * T_];     // 16 heads x 1024 keys, 64 KB (CDNA5: 320 KB/WGP)
  int bt = blockIdx.x;
  int b = bt >> 10, t = bt & 1023;
  int tid = threadIdx.x;

  // Async global->LDS fill of the 16x1024 logit row (GLOBAL_LOAD_ASYNC_TO_LDS_B128,
  // ASYNCcnt-tracked, no VGPR staging). LDS byte address = low 32 bits of the
  // generic shared pointer.
  for (int chunk = tid; chunk < (H_ * T_) / 4; chunk += 256) {
    int n = chunk >> 8;                 // 256 x 16B chunks per head row
    int s4 = (chunk & 255) * 4;
    const float* gsrc = logits + ((long long)(b * H_ + n) * T_ + t) * T_ + s4;
    u32 ldsaddr = (u32)(uintptr_t)(&row[chunk * 4]);
    asm volatile("global_load_async_to_lds_b128 %0, %1, off"
                 :: "v"(ldsaddr), "v"(gsrc) : "memory");
  }

  // overlap: per-t mixing parameters while the async copies are in flight
  float qw1[16], qw2[16], qdd[16], oqw1[16], oqw2[16], oqdd[16];
  long long pb = (long long)bt * 64;
#pragma unroll
  for (int n = 0; n < 16; ++n) {
    qw1[n]  = pw1[pb +        n];   // pre_q
    qw2[n]  = pw2[pb +        n];
    qdd[n]  = pdd[pb +        n];
    oqw1[n] = pw1[pb + 32 +   n];   // post_q
    oqw2[n] = pw2[pb + 32 +   n];
    oqdd[n] = pdd[pb + 32 +   n];
  }

#if __has_builtin(__builtin_amdgcn_s_wait_asynccnt)
  __builtin_amdgcn_s_wait_asynccnt(0);
#else
  asm volatile("s_wait_asynccnt 0x0" ::: "memory");
#endif
  __syncthreads();

  // pre-mix across heads, then causal mask
  for (int s = tid; s < T_; s += 256) {
    float v[16];
#pragma unroll
    for (int n = 0; n < 16; ++n) v[n] = row[n * T_ + s];
    long long kbo = (long long)(b * T_ + s) * 64 + 16;     // pre_k params at position s
    float dq = 0.f, dk = 0.f;
#pragma unroll
    for (int n = 0; n < 16; ++n) { dq += v[n] * qw1[n]; dk += v[n] * pw1[kbo + n]; }
#pragma unroll
    for (int m = 0; m < 16; ++m) {
      float o = v[m] + qw2[m] * dq + pw2[kbo + m] * dk
              + v[m] * qdd[m] + v[m] * pdd[kbo + m];
      if (s > t) o = NEGV;
      row[m * T_ + s] = o;
    }
  }
  __syncthreads();

  // per-head softmax: 8 waves, each wave owns heads {w, w+8}; wave32 shfl reductions
  int wave = tid >> 5, lane = tid & 31;
  for (int hh = wave; hh < H_; hh += 8) {
    float mx = -3.0e38f;
    for (int s = lane; s < T_; s += 32) mx = fmaxf(mx, row[hh * T_ + s]);
#pragma unroll
    for (int off = 16; off > 0; off >>= 1) mx = fmaxf(mx, __shfl_xor(mx, off, 32));
    float sum = 0.f;
    for (int s = lane; s < T_; s += 32) {
      float e = expf(row[hh * T_ + s] - mx);
      row[hh * T_ + s] = e;
      sum += e;
    }
#pragma unroll
    for (int off = 16; off > 0; off >>= 1) sum += __shfl_xor(sum, off, 32);
    float inv = 1.0f / sum;
    for (int s = lane; s < T_; s += 32) row[hh * T_ + s] *= inv;
  }
  __syncthreads();

  // post-mix on probabilities
  for (int s = tid; s < T_; s += 256) {
    float v[16];
#pragma unroll
    for (int n = 0; n < 16; ++n) v[n] = row[n * T_ + s];
    long long kbo = (long long)(b * T_ + s) * 64 + 48;     // post_k params at position s
    float dq = 0.f, dk = 0.f;
#pragma unroll
    for (int n = 0; n < 16; ++n) { dq += v[n] * oqw1[n]; dk += v[n] * pw1[kbo + n]; }
#pragma unroll
    for (int m = 0; m < 16; ++m)
      row[m * T_ + s] = v[m] + oqw2[m] * dq + pw2[kbo + m] * dk
                      + v[m] * oqdd[m] + v[m] * pdd[kbo + m];
  }
  __syncthreads();

  for (int idx = tid; idx < H_ * T_; idx += 256) {
    int n = idx >> 10, s = idx & 1023;
    probsB[((long long)(b * H_ + n) * T_ + t) * T_ + s] = f2bf(row[idx]);
  }
}

// ---------------- reshape o: [b,h,t,hd] f32 -> [b,t,h*hd] bf16 ----------------
__global__ void reshape_o_kernel(const float* __restrict__ o_tmp, u16* __restrict__ ob) {
  int idx = blockIdx.x * 256 + threadIdx.x;
  if (idx >= B_ * T_ * H_ * HD_) return;
  int hd = idx & 127;
  int t  = (idx >> 7) & 1023;
  int h  = (idx >> 17) & 15;
  int b  = idx >> 21;
  ob[((long long)(b * T_ + t) * H_ + h) * HD_ + hd] = f2bf(o_tmp[idx]);
}

// ---------------- host orchestration ----------------
extern "C" void kernel_launch(void* const* d_in, const int* in_sizes, int n_in,
                              void* d_out, int out_size, void* d_ws, size_t ws_size,
                              hipStream_t stream)
{
  (void)in_sizes; (void)n_in; (void)out_size; (void)ws_size;
  const float* x     = (const float*)d_in[0];
  /* d_in[1] = mask (bool) — causal, handled analytically */
  const float* freqs = (const float*)d_in[2];
  const float* wqkv  = (const float*)d_in[3];
  const float* wo    = (const float*)d_in[4];
  const float* dw1   = (const float*)d_in[5];
  const float* qkw   = (const float*)d_in[6];
  const float* ddw   = (const float*)d_in[7];
  float* out = (float*)d_out;

  const int BT = B_ * T_;              // 2048
  char* p = (char*)d_ws;
  auto alloc = [&](size_t bytes) -> char* {
    char* r = p;
    p += (bytes + 255) & ~(size_t)255;
    return r;
  };

  u16*   xb     = (u16*)  alloc((size_t)BT * D_ * 2);
  u16*   wqkvT  = (u16*)  alloc((size_t)(3 * H_ * HD_) * D_ * 2);
  u16*   woT    = (u16*)  alloc((size_t)D_ * (H_ * HD_) * 2);
  u16*   dw1T   = (u16*)  alloc((size_t)512 * D_ * 2);
  u16*   ddwT   = (u16*)  alloc((size_t)64 * D_ * 2);
  float* qkvf   = (float*)alloc((size_t)BT * (3 * H_ * HD_) * 4);
  u16*   qb     = (u16*)  alloc((size_t)BT * H_ * HD_ * 2);
  u16*   kb     = (u16*)  alloc((size_t)BT * H_ * HD_ * 2);
  u16*   vbT    = (u16*)  alloc((size_t)BT * H_ * HD_ * 2);
  float* hid    = (float*)alloc((size_t)BT * 512 * 4);
  float* ddpre  = (float*)alloc((size_t)BT * 64 * 4);
  float* pw1    = (float*)alloc((size_t)BT * 64 * 4);
  float* pw2    = (float*)alloc((size_t)BT * 64 * 4);
  float* pdd    = (float*)alloc((size_t)BT * 64 * 4);
  float* logits = (float*)alloc((size_t)B_ * H_ * T_ * T_ * 4);
  u16*   probsB = (u16*)  alloc((size_t)B_ * H_ * T_ * T_ * 2);
  float* o_tmp  = (float*)alloc((size_t)BT * H_ * HD_ * 4);
  u16*   ob     = (u16*)  alloc((size_t)BT * H_ * HD_ * 2);

  // 1) activations -> bf16
  f32_to_bf16_kernel<<<(BT * D_ + 255) / 256, 256, 0, stream>>>(x, xb, BT * D_);

  // 2) weight transposes (f32 -> bf16, [K,N] -> [N,K])
  transpose_bf16_kernel<<<dim3(6144 / 32, D_ / 32), 256, 0, stream>>>(wqkv, wqkvT, D_, 6144);
  transpose_bf16_kernel<<<dim3(D_   / 32, D_ / 32), 256, 0, stream>>>(wo,   woT,   D_, D_);
  transpose_bf16_kernel<<<dim3(512  / 32, D_ / 32), 256, 0, stream>>>(dw1,  dw1T,  D_, 512);
  transpose_bf16_kernel<<<dim3(64   / 32, D_ / 32), 256, 0, stream>>>(ddw,  ddwT,  D_, 64);

  // 3) qkv = x @ wqkv   (2048 x 6144 x 2048)
  gemm_bt_kernel<<<dim3(6144 / 256, BT / 32, 1), 128, 0, stream>>>(
      xb, wqkvT, qkvf, BT, 6144, D_, 0, 0, 0);

  // 4) dyn-weight projections: hid = x @ dw1 (N=512), ddpre = x @ dd_w (N=64)
  gemm_bt_kernel<<<dim3(2, BT / 32, 1), 128, 0, stream>>>(xb, dw1T, hid,   BT, 512, D_, 0, 0, 0);
  gemm_bt_kernel<<<dim3(1, BT / 32, 1), 128, 0, stream>>>(xb, ddwT, ddpre, BT, 64,  D_, 0, 0, 0);

  // 5) RoPE + pack
  rope_pack_kernel<<<(BT * H_ * 64) / 256, 256, 0, stream>>>(qkvf, freqs, qb, kb, vbT);

  // 6) dynamic mixing weights
  dynw_kernel<<<BT, 128, 0, stream>>>(hid, ddpre, qkw, pw1, pw2, pdd);

  // 7) logits[b,h] = q[b,h] @ k[b,h]^T   (32 x [1024 x 1024 x 128])
  gemm_bt_kernel<<<dim3(T_ / 256, T_ / 32, B_ * H_), 128, 0, stream>>>(
      qb, kb, logits, T_, T_, HD_,
      (long long)T_ * HD_, (long long)T_ * HD_, (long long)T_ * T_);

  // 8) pre-mix + mask + softmax + post-mix, one (b,t) row per block
  mixer_kernel<<<BT, 256, 0, stream>>>(logits, pw1, pw2, pdd, probsB);

  // 9) o[b,h] = probs[b,h] @ v[b,h]   (32 x [1024 x 128 x 1024])
  gemm_bt_kernel<<<dim3(1, T_ / 32, B_ * H_), 128, 0, stream>>>(
      probsB, vbT, o_tmp, T_, HD_, T_,
      (long long)T_ * T_, (long long)HD_ * T_, (long long)T_ * HD_);

  // 10) reshape to [b,t,h*hd] bf16
  reshape_o_kernel<<<(BT * D_) / 256, 256, 0, stream>>>(o_tmp, ob);

  // 11) out = o @ wo   (2048 x 2048 x 2048), f32 result
  gemm_bt_kernel<<<dim3(D_ / 256, BT / 32, 1), 128, 0, stream>>>(
      ob, woT, out, BT, D_, D_, 0, 0, 0);
}
